// T5Attention_67808943669376
// MI455X (gfx1250) — matrix operand order repaired and checked by reading, not compiled, not run on hardware
//
#include <hip/hip_runtime.h>
#include <hip/hip_bf16.h>

// ---------------------------------------------------------------------------
// T5 attention block for MI455X (gfx1250, wave32, WMMA).
//   B=4, Q=2048, D=1024, H=16, HD=64
// - All WMMA B-operands kept N-major in LDS -> fragment loads are 2x ds_load_b128.
// - GEMMs: 128x128 block tile, register-staged double-buffered LDS pipeline.
// - Attention: flash-style online softmax, WMMA for S=QK^T and O+=PV,
//   rel_bias (256 MB, the dominant HBM stream) loaded coalesced + prefetched.
// ---------------------------------------------------------------------------

typedef __attribute__((ext_vector_type(16))) _Float16 v16h;
typedef __attribute__((ext_vector_type(8)))  float    v8f;

#define B_   4
#define Q_   2048
#define D_   1024
#define H_   16
#define HD_  64
#define NH_  (H_ * HD_)   // 1024
#define M_   (B_ * Q_)    // 8192

union HV16 { uint4 u[2]; v16h h; };

// ---------------------------------------------------------------------------
// WMMA helpers (layouts per CDNA5 ISA 7.12.2, wave32)
// ---------------------------------------------------------------------------
__device__ inline v8f wmma16(v16h a, v16h b, v8f c) {
  return __builtin_amdgcn_wmma_f32_16x16x32_f16(
      false, a, false, b, (short)0, c, false, false);
}

// A fragment: 16x32 (MxK) from row-major source (stride in halves, mult of 8).
// lane&15 = M; lanes 0-15 take K {0..7,16..23}, lanes 16-31 K {8..15,24..31}.
__device__ inline v16h load_a_frag(const _Float16* __restrict__ s, int stride) {
  const int lane = threadIdx.x & 31;
  const _Float16* row = s + (lane & 15) * stride + ((lane >> 4) << 3);
  HV16 t;
  t.u[0] = *(const uint4*)(row);       // K kb+0..7
  t.u[1] = *(const uint4*)(row + 16);  // K kb+16..23
  return t.h;
}

// B fragment: 32x16 (KxN) from an N-MAJOR source sT[N][K] (stride mult of 8).
// lane&15 = N; lanes 0-15 hold K {0..7,16..23}, lanes 16-31 K {8..15,24..31}:
// per lane two 16-byte chunks -> 2x ds_load_b128.
__device__ inline v16h load_b_fragT(const _Float16* __restrict__ sT, int stride, int n0) {
  const int lane = threadIdx.x & 31;
  const _Float16* col = sT + (n0 + (lane & 15)) * stride + ((lane >> 4) << 3);
  HV16 t;
  t.u[0] = *(const uint4*)(col);
  t.u[1] = *(const uint4*)(col + 16);
  return t.h;
}

// ---------------------------------------------------------------------------
// f32 -> f16 conversion (plain)
// ---------------------------------------------------------------------------
__global__ void cvt_f32_f16(const float* __restrict__ s, _Float16* __restrict__ d, int n) {
  int i = blockIdx.x * blockDim.x + threadIdx.x;
  if (i < n) d[i] = (_Float16)s[i];
}

// f32 [1024][1024] -> f16 transposed [1024][1024]; LDS-tiled 32x32.
__global__ __launch_bounds__(256)
void cvt_transpose(const float* __restrict__ s, _Float16* __restrict__ d) {
  __shared__ float tile[32][33];
  const int t  = threadIdx.x;
  const int bx = blockIdx.x & 31;        // source col tile
  const int by = blockIdx.x >> 5;        // source row tile
  const int tc = t & 31, tr = t >> 5;    // 32 cols x 8 rows per pass
#pragma unroll
  for (int i = 0; i < 4; ++i) {
    int r = tr + i * 8;
    tile[r][tc] = s[(size_t)(by * 32 + r) * 1024 + bx * 32 + tc];
  }
  __syncthreads();
#pragma unroll
  for (int i = 0; i < 4; ++i) {
    int r = tr + i * 8;
    d[(size_t)(bx * 32 + r) * 1024 + by * 32 + tc] = (_Float16)tile[tc][r];
  }
}

// ---------------------------------------------------------------------------
// GEMM: C[m,n] = sum_k A[m,k] * Wt[n,k]   (Wt pre-transposed, f16, K=D_=1024)
// Block tile 128x128, 8 waves (4Mx2N), wave tile 32x64 (2x4 WMMA tiles).
// K-step 32, register-staged double-buffered LDS (one barrier / iteration).
// MODE 0: f16 out [B,H,Q,HD]        (Q and K projections)
// MODE 1: f16 out [B,H,HD,Q]        (V projection, transposed for P*V)
// MODE 2: f32 out row-major [M,NH]  (output projection)
// ---------------------------------------------------------------------------
template <int MODE>
__global__ __launch_bounds__(256)
void gemm_wmma(const _Float16* __restrict__ A, const _Float16* __restrict__ Wt,
               void* __restrict__ outp) {
  __shared__ _Float16 As [2][128 * 40];  // [m][k] 32 halves + pad 8
  __shared__ _Float16 BsT[2][128 * 40];  // [n][k] 32 halves + pad 8 (N-major)

  const int tid   = threadIdx.x;
  const int lane  = tid & 31;
  const int wave  = tid >> 5;
  const int waveM = wave >> 1;   // 0..3  (32 rows each)
  const int waveN = wave & 1;    // 0..1  (64 cols each)
  const int m0 = blockIdx.x * 128;
  const int n0 = blockIdx.y * 128;

  // staging indices: 512 uint4 chunks per 128x32 tile, 2 per thread per matrix
  const int r0 = (tid + 0)   >> 2, c0 = (tid + 0)   & 3;
  const int r1 = (tid + 256) >> 2, c1 = (tid + 256) & 3;

  v8f acc[2][4] = {};
  uint4 pa[2], pb[2];

  // preload k0 = 0
  pa[0] = *(const uint4*)(A  + (size_t)(m0 + r0) * D_ + c0 * 8);
  pa[1] = *(const uint4*)(A  + (size_t)(m0 + r1) * D_ + c1 * 8);
  pb[0] = *(const uint4*)(Wt + (size_t)(n0 + r0) * D_ + c0 * 8);
  pb[1] = *(const uint4*)(Wt + (size_t)(n0 + r1) * D_ + c1 * 8);
  *(uint4*)(&As [0][r0 * 40 + c0 * 8]) = pa[0];
  *(uint4*)(&As [0][r1 * 40 + c1 * 8]) = pa[1];
  *(uint4*)(&BsT[0][r0 * 40 + c0 * 8]) = pb[0];
  *(uint4*)(&BsT[0][r1 * 40 + c1 * 8]) = pb[1];

  for (int it = 0; it < D_ / 32; ++it) {
    __syncthreads();
    const int cur = it & 1;

    // issue next tile's global loads (overlaps the WMMA stream below)
    if (it + 1 < D_ / 32) {
      const int k0 = (it + 1) * 32;
      pa[0] = *(const uint4*)(A  + (size_t)(m0 + r0) * D_ + k0 + c0 * 8);
      pa[1] = *(const uint4*)(A  + (size_t)(m0 + r1) * D_ + k0 + c1 * 8);
      pb[0] = *(const uint4*)(Wt + (size_t)(n0 + r0) * D_ + k0 + c0 * 8);
      pb[1] = *(const uint4*)(Wt + (size_t)(n0 + r1) * D_ + k0 + c1 * 8);
    }

    // compute from current LDS buffer: 2 A-frags x 4 B-frags -> 8 WMMA
    v16h a0 = load_a_frag(&As[cur][(waveM * 32 + 0)  * 40], 40);
    v16h a1 = load_a_frag(&As[cur][(waveM * 32 + 16) * 40], 40);
#pragma unroll
    for (int tn = 0; tn < 4; ++tn) {
      v16h b = load_b_fragT(BsT[cur], 40, waveN * 64 + tn * 16);
      acc[0][tn] = wmma16(a0, b, acc[0][tn]);
      acc[1][tn] = wmma16(a1, b, acc[1][tn]);
    }

    // store next tile to the alternate buffer (disjoint from readers)
    if (it + 1 < D_ / 32) {
      const int nxt = cur ^ 1;
      *(uint4*)(&As [nxt][r0 * 40 + c0 * 8]) = pa[0];
      *(uint4*)(&As [nxt][r1 * 40 + c1 * 8]) = pa[1];
      *(uint4*)(&BsT[nxt][r0 * 40 + c0 * 8]) = pb[0];
      *(uint4*)(&BsT[nxt][r1 * 40 + c1 * 8]) = pb[1];
    }
  }

  // epilogue (straight-line per MODE; C row r lives in VGPR r / r+8 per half)
  const int halfsel = (lane >> 4) << 3;
#pragma unroll
  for (int tm = 0; tm < 2; ++tm)
#pragma unroll
    for (int tn = 0; tn < 4; ++tn)
#pragma unroll
      for (int r = 0; r < 8; ++r) {
        int gm = m0 + waveM * 32 + tm * 16 + r + halfsel;
        int gn = n0 + waveN * 64 + tn * 16 + (lane & 15);
        float v = acc[tm][tn][r];
        if (MODE == 2) {
          ((float*)outp)[(size_t)gm * NH_ + gn] = v;
        } else {
          int b = gm >> 11, q = gm & (Q_ - 1);
          int h = gn >> 6,  hd = gn & (HD_ - 1);
          if (MODE == 1)
            ((_Float16*)outp)[((size_t)(b * H_ + h) * HD_ + hd) * Q_ + q] = (_Float16)v;
          else
            ((_Float16*)outp)[((size_t)(b * H_ + h) * Q_ + q) * HD_ + hd] = (_Float16)v;
        }
      }
}

// ---------------------------------------------------------------------------
// Flash attention: one block = 64 q-rows of one (b,h); 4 waves x 16 rows.
// Qn: [B,H,Q,HD] f16  (A operand, row-major)
// Kn: [B,H,Q,HD] f16  (N-major for S = Q*K^T:  Ks[kcol][hd])
// Vt: [B,H,HD,Q] f16  (N-major for O += P*V:   Vs[hd][krow])
// bias: [H,Q,Q] f32   mask: [B,Q] f32   ctx out: [B*Q, H*HD] f16
// ---------------------------------------------------------------------------
__global__ __launch_bounds__(128)
void attn_wmma(const _Float16* __restrict__ Qn, const _Float16* __restrict__ Kn,
               const _Float16* __restrict__ Vt, const float* __restrict__ bias,
               const float* __restrict__ mask, _Float16* __restrict__ ctx) {
  __shared__ _Float16 Ks[64 * 72];      // [kcol][hd]  (N-major for S)
  __shared__ _Float16 Vs[64 * 72];      // [hd][krow]  (N-major for PV)
  __shared__ _Float16 Ps[4][16 * 72];   // per-wave P (row-major A operand)

  const int tid  = threadIdx.x;
  const int lane = tid & 31;
  const int w    = tid >> 5;
  const int bh   = blockIdx.x >> 5;          // (b*H + h)
  const int qt   = blockIdx.x & 31;
  const int b = bh >> 4, h = bh & 15;
  const int q0 = qt * 64;
  const int halfsel = (lane >> 4) << 3;

  const _Float16* Qb = Qn + ((size_t)bh * Q_ + q0 + w * 16) * HD_;
  const _Float16* Kb = Kn + (size_t)bh * Q_ * HD_;
  const _Float16* Vb = Vt + (size_t)bh * HD_ * Q_;

  // Q fragments persist across the K loop: 16 rows x 64 hd
  v16h qf0 = load_a_frag(Qb, HD_);        // hd 0..31
  v16h qf1 = load_a_frag(Qb + 32, HD_);   // hd 32..63

  v8f accO[4] = {};
  float mstat[8], lstat[8];
#pragma unroll
  for (int r = 0; r < 8; ++r) { mstat[r] = -1e30f; lstat[r] = 0.f; }

  const float scale = 0.125f;  // 1/sqrt(HD)

  for (int k0 = 0; k0 < Q_; k0 += 64) {
    __syncthreads();
    // stage K tile [kcol][hd] and V tile [hd][krow]: both contiguous uint4
#pragma unroll
    for (int c = 0; c < 4; ++c) {
      int idx = tid + c * 128;               // 0..511
      int row = idx >> 3, ch = idx & 7;
      *(uint4*)(&Ks[row * 72 + ch * 8]) =
          *(const uint4*)(Kb + (size_t)(k0 + row) * HD_ + ch * 8);
      *(uint4*)(&Vs[row * 72 + ch * 8]) =
          *(const uint4*)(Vb + (size_t)row * Q_ + k0 + ch * 8);
    }
    // prefetch next rel_bias tile (dominant HBM stream: 256 MB total)
    if (k0 + 64 < Q_)
      __builtin_prefetch(
          &bias[((size_t)h * Q_ + q0 + w * 16 + (lane >> 2)) * Q_ + k0 + 64 +
                (lane & 3) * 16], 0, 1);
    __syncthreads();

    // S = Q * K^T  (16 q-rows x 64 k-cols per wave)
    v8f S[4];
#pragma unroll
    for (int t = 0; t < 4; ++t) {
      v16h bk0 = load_b_fragT(Ks,      72, t * 16);  // hd 0..31
      v16h bk1 = load_b_fragT(Ks + 32, 72, t * 16);  // hd 32..63
      v8f s = {};
      s = wmma16(qf0, bk0, s);
      s = wmma16(qf1, bk1, s);
      S[t] = s;
    }

    // scale + rel_bias + mask (bias loads coalesced per half-wave row)
#pragma unroll
    for (int t = 0; t < 4; ++t) {
      int col = k0 + t * 16 + (lane & 15);
      float mv = mask[(size_t)b * Q_ + col];
#pragma unroll
      for (int r = 0; r < 8; ++r) {
        int qrow = q0 + w * 16 + r + halfsel;
        float bv = bias[((size_t)h * Q_ + qrow) * Q_ + col];
        S[t][r] = S[t][r] * scale + bv + mv;
      }
    }

    // online softmax: row stats within 16-lane halves (xor 1,2,4,8)
    float mnew[8], alpha[8];
#pragma unroll
    for (int r = 0; r < 8; ++r) {
      float mx = S[0][r];
#pragma unroll
      for (int t = 1; t < 4; ++t) mx = fmaxf(mx, S[t][r]);
#pragma unroll
      for (int off = 1; off < 16; off <<= 1)
        mx = fmaxf(mx, __shfl_xor(mx, off, 32));
      mnew[r]  = fmaxf(mstat[r], mx);
      alpha[r] = __expf(mstat[r] - mnew[r]);
      mstat[r] = mnew[r];
    }

#pragma unroll
    for (int r = 0; r < 8; ++r) {
      float rs = 0.f;
#pragma unroll
      for (int t = 0; t < 4; ++t) {
        float p = __expf(S[t][r] - mnew[r]);
        S[t][r] = p;
        rs += p;
      }
#pragma unroll
      for (int off = 1; off < 16; off <<= 1)
        rs += __shfl_xor(rs, off, 32);
      lstat[r] = lstat[r] * alpha[r] + rs;
#pragma unroll
      for (int t = 0; t < 4; ++t) accO[t][r] *= alpha[r];
    }

    // P: C-layout -> LDS (row-major) -> A-layout
#pragma unroll
    for (int t = 0; t < 4; ++t)
#pragma unroll
      for (int r = 0; r < 8; ++r)
        Ps[w][(r + halfsel) * 72 + t * 16 + (lane & 15)] = (_Float16)S[t][r];

    __syncthreads();

    v16h p0 = load_a_frag(&Ps[w][0], 72);        // k-cols 0..31
    v16h p1 = load_a_frag(&Ps[w][0] + 32, 72);   // k-cols 32..63
#pragma unroll
    for (int t = 0; t < 4; ++t) {
      v16h bv0 = load_b_fragT(Vs,      72, t * 16);  // k-rows 0..31
      v16h bv1 = load_b_fragT(Vs + 32, 72, t * 16);  // k-rows 32..63
      accO[t] = wmma16(p0, bv0, accO[t]);
      accO[t] = wmma16(p1, bv1, accO[t]);
    }
  }

  // normalize and write context [B*Q, H*HD]
#pragma unroll
  for (int r = 0; r < 8; ++r) {
    float inv = 1.f / lstat[r];
    int qrow = q0 + w * 16 + r + halfsel;
#pragma unroll
    for (int t = 0; t < 4; ++t) {
      int hd = t * 16 + (lane & 15);
      ctx[((size_t)b * Q_ + qrow) * NH_ + h * HD_ + hd] =
          (_Float16)(accO[t][r] * inv);
    }
  }
}

// ---------------------------------------------------------------------------
// Host launcher
// ---------------------------------------------------------------------------
extern "C" void kernel_launch(void* const* d_in, const int* in_sizes, int n_in,
                              void* d_out, int out_size, void* d_ws, size_t ws_size,
                              hipStream_t stream) {
  (void)in_sizes; (void)n_in; (void)out_size; (void)ws_size;

  const float* x    = (const float*)d_in[0];
  const float* mask = (const float*)d_in[1];
  const float* bias = (const float*)d_in[2];   // (1,H,Q,Q)
  const float* Wq   = (const float*)d_in[3];
  const float* Wk   = (const float*)d_in[4];
  const float* Wv   = (const float*)d_in[5];
  const float* Wo   = (const float*)d_in[6];

  char* ws = (char*)d_ws;
  size_t off = 0;
  auto alloc = [&](size_t bytes) -> char* {
    char* p = ws + off;
    off += (bytes + 255) & ~(size_t)255;
    return p;
  };
  _Float16* xh  = (_Float16*)alloc((size_t)M_ * D_  * sizeof(_Float16));
  _Float16* wqt = (_Float16*)alloc((size_t)D_ * NH_ * sizeof(_Float16));  // [n][k]
  _Float16* wkt = (_Float16*)alloc((size_t)D_ * NH_ * sizeof(_Float16));
  _Float16* wvt = (_Float16*)alloc((size_t)D_ * NH_ * sizeof(_Float16));
  _Float16* wot = (_Float16*)alloc((size_t)NH_ * D_ * sizeof(_Float16));
  _Float16* Qn  = (_Float16*)alloc((size_t)M_ * NH_ * sizeof(_Float16));  // [B,H,Q,HD]
  _Float16* Kn  = (_Float16*)alloc((size_t)M_ * NH_ * sizeof(_Float16));  // [B,H,Q,HD]
  _Float16* Vt  = (_Float16*)alloc((size_t)M_ * NH_ * sizeof(_Float16));  // [B,H,HD,Q]
  _Float16* ctx = (_Float16*)alloc((size_t)M_ * NH_ * sizeof(_Float16));

  const int nx = M_ * D_;
  cvt_f32_f16<<<(nx + 255) / 256, 256, 0, stream>>>(x, xh, nx);
  cvt_transpose<<<1024, 256, 0, stream>>>(Wq, wqt);
  cvt_transpose<<<1024, 256, 0, stream>>>(Wk, wkt);
  cvt_transpose<<<1024, 256, 0, stream>>>(Wv, wvt);
  cvt_transpose<<<1024, 256, 0, stream>>>(Wo, wot);

  dim3 gg(M_ / 128, NH_ / 128), gb(256);
  gemm_wmma<0><<<gg, gb, 0, stream>>>(xh, wqt, (void*)Qn);
  gemm_wmma<0><<<gg, gb, 0, stream>>>(xh, wkt, (void*)Kn);
  gemm_wmma<1><<<gg, gb, 0, stream>>>(xh, wvt, (void*)Vt);

  attn_wmma<<<B_ * H_ * (Q_ / 64), 128, 0, stream>>>(Qn, Kn, Vt, bias, mask, ctx);

  gemm_wmma<2><<<gg, gb, 0, stream>>>(ctx, wot, d_out);
}
